// Conttime_73761768342073
// MI455X (gfx1250) — matrix operand order repaired and checked by reading, not compiled, not run on hardware
//
#include <hip/hip_runtime.h>
#include <hip/hip_bf16.h>

// ---------------------------------------------------------------------------
// Continuous-time LSTM (neural Hawkes) scan for MI455X / gfx1250.
// B=512, S=512, T=511, HID=256, GATES=7*256=1792, K=2*HID=512, N_TYPES=128.
// ---------------------------------------------------------------------------

typedef __attribute__((ext_vector_type(16))) __bf16 v16bf;
typedef __attribute__((ext_vector_type(8)))  __bf16 v8bf;
typedef __attribute__((ext_vector_type(8)))  float  v8f;

#define CT_B     512
#define CT_S     512
#define CT_T     511
#define CT_H     256
#define CT_G     1792          // 7*H
#define CT_K     512           // 2*H
#define CT_NT    128
#define NBLK_SCAN 112          // persistent blocks (must all be resident)
#define NTHR      256          // 8 waves per block
#define NWAVES    (NBLK_SCAN * 8)   // 896 = 16 * 56 tile-blocks of 32x32

// workspace layout (bytes)
#define WS_BAR    ((size_t)0)                              // 256 B barrier block
#define WS_WCT    ((size_t)256)                            // bf16 W_cell^T [1792][512]
#define WS_WLT    (WS_WCT  + (size_t)CT_G * CT_K * 2)      // bf16 W_lam^T  [128][256]
#define WS_HBF    (WS_WLT  + (size_t)CT_NT * CT_H * 2)     // bf16 h state  [512][256]
#define WS_C      (WS_HBF  + (size_t)CT_B * CT_H * 2)      // f32 c state
#define WS_CBAR   (WS_C    + (size_t)CT_B * CT_H * 4)      // f32 c_bar state
#define WS_GATES  (WS_CBAR + (size_t)CT_B * CT_H * 4)      // f32 gates scratch [512][1792]

// d_out layout (floats): intensity (B,T,128) then h,c,cbar,decay,gate each (T,B,H)
#define OUT_INT_SZ ((size_t)CT_B * CT_T * CT_NT)           // 33,488,896
#define OUT_BLK_SZ ((size_t)CT_T * CT_B * CT_H)            // 66,977,792

__device__ __forceinline__ float sigmoidf_(float x) {
  return 1.0f / (1.0f + __expf(-x));
}
__device__ __forceinline__ float softplusf_(float x) {
  // numerically stable softplus, BETA = 1
  return fmaxf(x, 0.0f) + log1pf(__expf(-fabsf(x)));
}
__device__ __forceinline__ v16bf cvt_pair(v8f lo, v8f hi) {
  v16bf r;
#pragma unroll
  for (int i = 0; i < 8; ++i) { r[i] = (__bf16)lo[i]; r[i + 8] = (__bf16)hi[i]; }
  return r;
}
__device__ __forceinline__ v16bf wmma_bf16(v16bf a, v16bf b, v8f) = delete; // guard

// device-wide split barrier (sense reversal). All NBLK_SCAN blocks resident.
__device__ __forceinline__ void grid_sync(volatile unsigned* cnt, volatile unsigned* sns) {
  __syncthreads();
  if (threadIdx.x == 0) {
    unsigned my = *sns;
    __threadfence();
    unsigned prev = atomicAdd((unsigned*)cnt, 1u);
    if (prev == (unsigned)(NBLK_SCAN - 1)) {
      *cnt = 0u;
      __threadfence();
      atomicAdd((unsigned*)sns, 1u);
    } else {
      while (*sns == my) { __builtin_amdgcn_s_sleep(2); }
    }
  }
  __threadfence();
  __syncthreads();
}

// ---------------------------------------------------------------------------
// init: zero barrier + states, build bf16 transposed weights
// ---------------------------------------------------------------------------
__global__ void __launch_bounds__(NTHR) k_ct_init(const float* __restrict__ Wc,
                                                  const float* __restrict__ Wl,
                                                  char* __restrict__ ws) {
  const int tid = blockIdx.x * blockDim.x + threadIdx.x;
  const int nth = gridDim.x * blockDim.x;
  unsigned* bar = (unsigned*)(ws + WS_BAR);
  __bf16* WcT  = (__bf16*)(ws + WS_WCT);
  __bf16* WlT  = (__bf16*)(ws + WS_WLT);
  __bf16* hBf  = (__bf16*)(ws + WS_HBF);
  float*  cS   = (float*)(ws + WS_C);
  float*  cbS  = (float*)(ws + WS_CBAR);

  if (tid < 64) bar[tid] = 0u;
  for (int i = tid; i < CT_G * CT_K; i += nth) {       // WcT[n][k] = Wc[k][n]
    int n = i / CT_K, k = i % CT_K;
    WcT[i] = (__bf16)Wc[(size_t)k * CT_G + n];
  }
  for (int i = tid; i < CT_NT * CT_H; i += nth) {      // WlT[n][k] = Wl[k][n]
    int n = i / CT_H, k = i % CT_H;
    WlT[i] = (__bf16)Wl[(size_t)k * CT_NT + n];
  }
  for (int i = tid; i < CT_B * CT_H; i += nth) {
    hBf[i] = (__bf16)0.0f; cS[i] = 0.0f; cbS[i] = 0.0f;
  }
}

// ---------------------------------------------------------------------------
// persistent scan kernel: per step t a 512x1792 (K=512) bf16 WMMA GEMM,
// grid barrier, elementwise CT-LSTM update, grid barrier.
// ---------------------------------------------------------------------------
__global__ void __launch_bounds__(NTHR, 1) k_ct_scan(const int* __restrict__ types,
                                                     const float* __restrict__ dtime,
                                                     const float* __restrict__ emb,
                                                     const float* __restrict__ bcell,
                                                     char* __restrict__ ws,
                                                     float* __restrict__ out) {
  volatile unsigned* barCnt = (volatile unsigned*)(ws + WS_BAR);
  volatile unsigned* barSns = barCnt + 1;
  const __bf16* WcT = (const __bf16*)(ws + WS_WCT);
  __bf16* hBf = (__bf16*)(ws + WS_HBF);
  float*  cS  = (float*)(ws + WS_C);
  float*  cbS = (float*)(ws + WS_CBAR);
  float*  gts = (float*)(ws + WS_GATES);

  const int lane = threadIdx.x & 31;
  const int wid  = blockIdx.x * 8 + (threadIdx.x >> 5);  // 0..895
  const int rt   = wid / 56;                             // 16 row-blocks of 32
  const int cb   = wid % 56;                             // 56 col-blocks of 32
  const int r0   = rt * 32;
  const int n0   = cb * 32;
  const int mA0  = r0 + (lane & 15);                     // A rows (two 16-row tiles)
  const int mA1  = mA0 + 16;
  const int rel0 = (lane & 16) ? 8  : 0;                 // A-layout K chunk offset
  const int bko  = (lane & 16) ? 8  : 0;                 // B-layout K offset
  const int nc0  = n0 + (lane & 15);                     // C/B columns (two tiles)
  const int nc1  = nc0 + 16;
  const int rwr  = ((lane >> 4) << 3);                   // C-layout row base in tile

  float* outH  = out + OUT_INT_SZ;
  float* outC  = outH  + OUT_BLK_SZ;
  float* outCB = outC  + OUT_BLK_SZ;
  float* outD  = outCB + OUT_BLK_SZ;
  float* outO  = outD  + OUT_BLK_SZ;

  const float bias0 = bcell[nc0];
  const float bias1 = bcell[nc1];

  for (int t = 0; t < CT_T; ++t) {
    // ---- phase 1: gates = [emb_t | h] @ W_cell  (bf16 WMMA, f32 accum) ----
    const float* e0 = emb + (size_t)types[mA0 * CT_S + t] * CT_H;
    const float* e1 = emb + (size_t)types[mA1 * CT_S + t] * CT_H;

    v8f acc00 = {}, acc01 = {}, acc10 = {}, acc11 = {};
#pragma unroll 4
    for (int kk = 0; kk < 16; ++kk) {
      const int k = kk * 32;
      v16bf a0, a1;
      if (k < CT_H) {                                    // embedding half (fp32 gather)
        a0 = cvt_pair(*(const v8f*)(e0 + k + rel0), *(const v8f*)(e0 + k + rel0 + 16));
        a1 = cvt_pair(*(const v8f*)(e1 + k + rel0), *(const v8f*)(e1 + k + rel0 + 16));
      } else {                                           // recurrent half (bf16 h)
        const __bf16* h0 = hBf + (size_t)mA0 * CT_H + (k - CT_H) + rel0;
        const __bf16* h1 = hBf + (size_t)mA1 * CT_H + (k - CT_H) + rel0;
        v8bf x0 = *(const v8bf*)h0, x1 = *(const v8bf*)(h0 + 16);
        v8bf y0 = *(const v8bf*)h1, y1 = *(const v8bf*)(h1 + 16);
        a0 = __builtin_shufflevector(x0, x1, 0,1,2,3,4,5,6,7,8,9,10,11,12,13,14,15);
        a1 = __builtin_shufflevector(y0, y1, 0,1,2,3,4,5,6,7,8,9,10,11,12,13,14,15);
      }
      const __bf16* w0 = WcT + (size_t)nc0 * CT_K + k + bko;
      const __bf16* w1 = WcT + (size_t)nc1 * CT_K + k + bko;
      v16bf b0 = __builtin_shufflevector(*(const v8bf*)w0, *(const v8bf*)(w0 + 16), 0,1,2,3,4,5,6,7,8,9,10,11,12,13,14,15);
      v16bf b1 = __builtin_shufflevector(*(const v8bf*)w1, *(const v8bf*)(w1 + 16), 0,1,2,3,4,5,6,7,8,9,10,11,12,13,14,15);
      acc00 = __builtin_amdgcn_wmma_f32_16x16x32_bf16(false, a0, false, b0, (short)0, acc00, false, false);
      acc01 = __builtin_amdgcn_wmma_f32_16x16x32_bf16(false, a0, false, b1, (short)0, acc01, false, false);
      acc10 = __builtin_amdgcn_wmma_f32_16x16x32_bf16(false, a1, false, b0, (short)0, acc10, false, false);
      acc11 = __builtin_amdgcn_wmma_f32_16x16x32_bf16(false, a1, false, b1, (short)0, acc11, false, false);
    }
#pragma unroll
    for (int j = 0; j < 8; ++j) {
      gts[(size_t)(r0 + rwr + j)       * CT_G + nc0] = acc00[j] + bias0;
      gts[(size_t)(r0 + rwr + j)       * CT_G + nc1] = acc01[j] + bias1;
      gts[(size_t)(r0 + 16 + rwr + j)  * CT_G + nc0] = acc10[j] + bias0;
      gts[(size_t)(r0 + 16 + rwr + j)  * CT_G + nc1] = acc11[j] + bias1;
    }
    grid_sync(barCnt, barSns);

    // ---- phase 2: elementwise CT-LSTM state update ----
    const size_t tOff = (size_t)t * (CT_B * CT_H);
    for (int idx = blockIdx.x * NTHR + threadIdx.x; idx < CT_B * CT_H;
         idx += NBLK_SCAN * NTHR) {
      const int b = idx >> 8, j = idx & 255;
      const float* g = gts + (size_t)b * CT_G + j;
      const float i_   = sigmoidf_(g[0]);
      const float f_   = sigmoidf_(g[256]);
      const float z_   = tanhf(g[512]);
      const float o_   = sigmoidf_(g[768]);
      const float ib_  = sigmoidf_(g[1024]);
      const float fb_  = sigmoidf_(g[1280]);
      const float dlt  = softplusf_(g[1536]);
      const float dt   = dtime[(size_t)b * CT_S + t + 1];
      const float c_i  = f_ * cS[idx] + i_ * z_;
      const float cbn  = fb_ * cbS[idx] + ib_ * z_;
      const float c_t  = cbn + (c_i - cbn) * __expf(-dlt * dt);
      const float h_n  = o_ * tanhf(c_t);
      outH [tOff + idx] = h_n;
      outC [tOff + idx] = c_i;
      outCB[tOff + idx] = cbn;
      outD [tOff + idx] = dlt;
      outO [tOff + idx] = o_;
      cS[idx]  = c_t;
      cbS[idx] = cbn;
      hBf[idx] = (__bf16)h_n;
    }
    grid_sync(barCnt, barSns);
  }
}

// ---------------------------------------------------------------------------
// intensity = softplus(h_out @ W_lam + b_lam), written transposed to (B,T,128)
// one wave per 16 rows x 128 cols, K=256 bf16 WMMA
// ---------------------------------------------------------------------------
__global__ void __launch_bounds__(NTHR) k_ct_intensity(const float* __restrict__ blam,
                                                       char* __restrict__ ws,
                                                       float* __restrict__ out) {
  const __bf16* WlT = (const __bf16*)(ws + WS_WLT);
  const float* outH = out + OUT_INT_SZ;

  const int lane = threadIdx.x & 31;
  const int wid  = blockIdx.x * 8 + (threadIdx.x >> 5);   // 0..16351
  const int m0   = wid * 16;
  const int mA   = m0 + (lane & 15);
  const int rel0 = (lane & 16) ? 8  : 0;
  const int bko  = (lane & 16) ? 8  : 0;
  const int rwr  = ((lane >> 4) << 3);
  const float* hrow = outH + (size_t)mA * CT_H;

  v8f acc[8] = {};
#pragma unroll
  for (int kk = 0; kk < 8; ++kk) {
    const int k = kk * 32;
    v16bf a = cvt_pair(*(const v8f*)(hrow + k + rel0),
                       *(const v8f*)(hrow + k + rel0 + 16));
#pragma unroll
    for (int tI = 0; tI < 8; ++tI) {
      const int n = tI * 16 + (lane & 15);
      const __bf16* wp = WlT + (size_t)n * CT_H + k + bko;
      v16bf b = __builtin_shufflevector(*(const v8bf*)wp, *(const v8bf*)(wp + 16), 0,1,2,3,4,5,6,7,8,9,10,11,12,13,14,15);
      acc[tI] = __builtin_amdgcn_wmma_f32_16x16x32_bf16(false, a, false, b, (short)0, acc[tI], false, false);
    }
  }
#pragma unroll
  for (int tI = 0; tI < 8; ++tI) {
    const int col = tI * 16 + (lane & 15);
    const float bl = blam[col];
#pragma unroll
    for (int j = 0; j < 8; ++j) {
      const int m  = m0 + rwr + j;
      const int tt = m >> 9;          // m / 512  (B=512)
      const int bb = m & 511;
      out[(size_t)bb * (CT_T * CT_NT) + (size_t)tt * CT_NT + col] =
          softplusf_(acc[tI][j] + bl);
    }
  }
}

// ---------------------------------------------------------------------------
extern "C" void kernel_launch(void* const* d_in, const int* in_sizes, int n_in,
                              void* d_out, int out_size, void* d_ws, size_t ws_size,
                              hipStream_t stream) {
  const int*   types = (const int*)d_in[0];
  const float* dtime = (const float*)d_in[1];
  const float* emb   = (const float*)d_in[2];
  const float* Wc    = (const float*)d_in[3];
  const float* bc    = (const float*)d_in[4];
  const float* Wl    = (const float*)d_in[5];
  const float* bl    = (const float*)d_in[6];
  char*  ws  = (char*)d_ws;
  float* out = (float*)d_out;

  k_ct_init<<<512, NTHR, 0, stream>>>(Wc, Wl, ws);
  k_ct_scan<<<NBLK_SCAN, NTHR, 0, stream>>>(types, dtime, emb, bc, ws, out);
  // (T*B/16) / 8 waves-per-block = 2044 blocks
  k_ct_intensity<<<2044, NTHR, 0, stream>>>(bl, ws, out);
}
